// GatedDeltaNet_10015863734534
// MI455X (gfx1250) — compile-verified, matrix operand order checked by reading, not run on hardware
//
#include <hip/hip_runtime.h>
#include <hip/hip_bf16.h>

// ---------------------------------------------------------------------------
// Gated DeltaNet forward for MI455X (gfx1250, wave32, WMMA).
// B=2 T=4096 H=16 DK=DV=128 BT=64 NT=64, conv K=4.
// ---------------------------------------------------------------------------

#define DEV static __device__ __forceinline__

typedef __attribute__((ext_vector_type(16))) _Float16 v16h;
typedef __attribute__((ext_vector_type(8)))  float    v8f;

constexpr int Bc = 2, Tc = 4096, Hc = 16, DKc = 128, DVc = 128;
constexpr int BTc = 64, NTc = 64, KCc = 4, VBc = 32;
constexpr float SCALEc = 0.08838834764831845f;   // 128^-0.5
constexpr float L2EPS  = 1e-6f;
constexpr float EPSc   = 1e-6f;

// K index inside a v16h WMMA operand (16x16x32 f16):
// element e, lane-half hh (lane>>4): K = 16*(e/8) + 8*hh + (e%8)
DEV int kmap(int e, int hh) { return ((e >> 3) << 4) + (hh << 3) + (e & 7); }

// A operand: rows m0..m0+15 (M from lane&15), K-cols k0..k0+31, row-major f16 src
DEV v16h load_a_rm(const _Float16* s, int ld, int m0, int k0, int m, int hh) {
  v16h a;
#pragma unroll
  for (int e = 0; e < 16; ++e) a[e] = s[(m0 + m) * ld + k0 + kmap(e, hh)];
  return a;
}
// A operand from f32 row-major src (e.g. Tm)
DEV v16h load_a_f32(const float* s, int ld, int m0, int k0, int m, int hh) {
  v16h a;
#pragma unroll
  for (int e = 0; e < 16; ++e) a[e] = (_Float16)s[(m0 + m) * ld + k0 + kmap(e, hh)];
  return a;
}
// B operand: K-major f16 src [K][ld], cols n0..n0+15 (N from lane&15)
DEV v16h load_b_km(const _Float16* s, int ld, int k0, int n0, int n, int hh) {
  v16h b;
#pragma unroll
  for (int e = 0; e < 16; ++e) b[e] = s[(k0 + kmap(e, hh)) * ld + n0 + n];
  return b;
}
// B operand: N-major f16 src [N][ld] (i.e. B = srcT), used for Q@K^T style
DEV v16h load_b_nm(const _Float16* s, int ld, int n0, int k0, int n, int hh) {
  v16h b;
#pragma unroll
  for (int e = 0; e < 16; ++e) b[e] = s[(n0 + n) * ld + k0 + kmap(e, hh)];
  return b;
}
// B operand from f32 K-major src (state S)
DEV v16h load_b_km_f32(const float* s, int ld, int k0, int n0, int n, int hh) {
  v16h b;
#pragma unroll
  for (int e = 0; e < 16; ++e) b[e] = (_Float16)s[(k0 + kmap(e, hh)) * ld + n0 + n];
  return b;
}

DEV v8f wmma16(v16h a, v16h b, v8f c) {
  // D = A*B + C ; fp16 in, fp32 accumulate
  return __builtin_amdgcn_wmma_f32_16x16x32_f16(false, a, false, b, (short)0, c,
                                                false, false);
}

DEV float sigmoidf_(float x) { return 1.0f / (1.0f + __expf(-x)); }

// ---------------------------------------------------------------------------
// Kernel 1: causal depthwise conv1d (K=4) + SiLU, optional l2norm (+scale),
// reshape (B, H*D, T) -> heads (B*H, T, D) in f16.
// mode: 0 = q (l2norm * DK^-0.5), 1 = k (l2norm), 2 = v (plain)
// grid (B*H, NT), block 256
// ---------------------------------------------------------------------------
__global__ void conv_silu_kernel(const float* __restrict__ x,
                                 const float* __restrict__ w,
                                 const float* __restrict__ bias,
                                 _Float16* __restrict__ out, int mode) {
  __shared__ float sy[BTc * DKc];
  __shared__ float ss[BTc];
  const int bh = blockIdx.x, b = bh / Hc, h = bh % Hc;
  const int t0 = blockIdx.y * BTc, tid = threadIdx.x;

  for (int idx = tid; idx < BTc * DKc; idx += 256) {
    const int tl = idx >> 7, d = idx & 127, t = t0 + tl;
    const int c = h * DKc + d;
    const size_t rb = ((size_t)(b * Hc * DKc + c)) * Tc;
    float acc = bias[c];
#pragma unroll
    for (int i = 0; i < KCc; ++i) {
      const int tt = t - (KCc - 1) + i;
      if (tt >= 0) acc += w[c * KCc + i] * x[rb + tt];
    }
    sy[idx] = acc * sigmoidf_(acc);  // SiLU
  }
  __syncthreads();
  if (mode < 2 && tid < BTc) {
    float s = 0.f;
#pragma unroll 4
    for (int d = 0; d < DKc; ++d) { float v = sy[tid * DKc + d]; s += v * v; }
    float inv = 1.0f / sqrtf(s + L2EPS);
    ss[tid] = (mode == 0) ? inv * SCALEc : inv;
  }
  __syncthreads();
  for (int idx = tid; idx < BTc * DKc; idx += 256) {
    const int tl = idx >> 7, d = idx & 127;
    const float v = sy[idx] * ((mode < 2) ? ss[tl] : 1.0f);
    out[((size_t)bh * Tc + (t0 + tl)) * DKc + d] = (_Float16)v;
  }
}

// ---------------------------------------------------------------------------
// Kernel 2: per-chunk prep (fully parallel over B*H*NT chunks).
//   A  = -(k_beta @ k^T) * decay, strictly lower
//   Tm = (I - A)^-1 (forward substitution, unit lower triangular)
//   w  = Tm @ (k_beta * exp(gcum))   -> wbuf (f16)
//   u  = Tm @ (v * beta)             -> ubuf (f16)
// grid B*H*NT, block 256 (8 waves)
// ---------------------------------------------------------------------------
__global__ void prep_kernel(const _Float16* __restrict__ kh,
                            const _Float16* __restrict__ vh,
                            const float* __restrict__ g,
                            const float* __restrict__ beta,
                            _Float16* __restrict__ wbuf,
                            _Float16* __restrict__ ubuf) {
  __shared__ _Float16 sk[BTc * DKc];   // k chunk
  __shared__ _Float16 sx[BTc * DKc];   // k*beta -> k*beta*exp(gc) -> v*beta
  __shared__ float    sA[BTc * BTc];   // A, then Tm (in place)
  __shared__ float    srow[BTc];
  __shared__ float    sgl[BTc], sgc[BTc], sbt[BTc];
  __shared__ float    segc[BTc], sinv[BTc];   // e^{gc}, e^{-gc}

  const int bh = blockIdx.x / NTc, nc = blockIdx.x % NTc, t0 = nc * BTc;
  const int tid = threadIdx.x, lane = tid & 31, wid = tid >> 5;
  const int m = lane & 15, hh = lane >> 4;
  const size_t base = ((size_t)bh * Tc + t0) * DKc;

  for (int idx = tid; idx < BTc * DKc; idx += 256) sk[idx] = kh[base + idx];
  if (tid < BTc) {
    sgl[tid] = g[(size_t)bh * Tc + t0 + tid];
    sbt[tid] = beta[(size_t)bh * Tc + t0 + tid];
  }
  __syncthreads();
  if (tid < BTc) {
    float s = 0.f;
    for (int j = 0; j <= tid; ++j) s += sgl[j];
    sgc[tid] = s;
    segc[tid] = __expf(s);
    sinv[tid] = __expf(-s);
  }
  __syncthreads();
  for (int idx = tid; idx < BTc * DKc; idx += 256) {
    const int tl = idx >> 7;
    sx[idx] = (_Float16)((float)sk[idx] * sbt[tl]);
  }
  __syncthreads();

  // A = -(k_beta @ k^T) * decay, strictly lower (16 tiles; 2 per wave)
  for (int rep = 0; rep < 2; ++rep) {
    const int tile = wid + 8 * rep;
    const int i0 = (tile >> 2) * 16, j0 = (tile & 3) * 16;
    v8f c = {};
#pragma unroll
    for (int ks = 0; ks < 4; ++ks) {
      v16h a = load_a_rm(sx, DKc, i0, ks * 32, m, hh);
      v16h b = load_b_nm(sk, DKc, j0, ks * 32, m, hh);
      c = wmma16(a, b, c);
    }
#pragma unroll
    for (int r = 0; r < 8; ++r) {
      const int i = i0 + r + 8 * hh, j = j0 + m;
      const float v = -c[r] * segc[i] * sinv[j];   // e^{gc_i - gc_j}
      sA[i * BTc + j] = (i > j) ? v : 0.0f;
    }
  }
  __syncthreads();

  // Tm = (I - A)^-1 in place: X[i][:] = e_i + sum_{l<i} A[i][l] * X[l][:]
  for (int i = 0; i < BTc; ++i) {
    if (tid < BTc) srow[tid] = sA[i * BTc + tid];
    __syncthreads();
    if (tid < BTc) {
      float acc = (tid == i) ? 1.0f : 0.0f;
      for (int l = 0; l < i; ++l) acc += srow[l] * sA[l * BTc + tid];
      sA[i * BTc + tid] = acc;
    }
    __syncthreads();
  }

  // k_beta * exp(gcum)
  for (int idx = tid; idx < BTc * DKc; idx += 256) {
    const int tl = idx >> 7;
    sx[idx] = (_Float16)((float)sx[idx] * segc[tl]);
  }
  __syncthreads();

  // w = Tm @ kbg  (32 tiles; 4 per wave)
  for (int rep = 0; rep < 4; ++rep) {
    const int tile = wid + 8 * rep;
    const int i0 = (tile >> 3) * 16, n0 = (tile & 7) * 16;
    v8f c = {};
#pragma unroll
    for (int ks = 0; ks < 2; ++ks) {
      v16h a = load_a_f32(sA, BTc, i0, ks * 32, m, hh);
      v16h b = load_b_km(sx, DKc, ks * 32, n0, m, hh);
      c = wmma16(a, b, c);
    }
#pragma unroll
    for (int r = 0; r < 8; ++r) {
      const int row = t0 + i0 + r + 8 * hh, col = n0 + m;
      wbuf[((size_t)bh * Tc + row) * DKc + col] = (_Float16)c[r];
    }
  }
  __syncthreads();

  // v * beta
  for (int idx = tid; idx < BTc * DVc; idx += 256) {
    const int tl = idx >> 7;
    sx[idx] = (_Float16)((float)vh[base + idx] * sbt[tl]);
  }
  __syncthreads();

  // u = Tm @ v_beta
  for (int rep = 0; rep < 4; ++rep) {
    const int tile = wid + 8 * rep;
    const int i0 = (tile >> 3) * 16, n0 = (tile & 7) * 16;
    v8f c = {};
#pragma unroll
    for (int ks = 0; ks < 2; ++ks) {
      v16h a = load_a_f32(sA, BTc, i0, ks * 32, m, hh);
      v16h b = load_b_km(sx, DVc, ks * 32, n0, m, hh);
      c = wmma16(a, b, c);
    }
#pragma unroll
    for (int r = 0; r < 8; ++r) {
      const int row = t0 + i0 + r + 8 * hh, col = n0 + m;
      ubuf[((size_t)bh * Tc + row) * DVc + col] = (_Float16)c[r];
    }
  }
}

// ---------------------------------------------------------------------------
// Kernel 3: sequential chunk scan. Grid = B*H * (DV/VB) (VB=32 columns of V/S
// per block -> 128 independent workgroups). S (128 x 32 f32) lives in LDS.
// Per chunk:
//   v_new = u - w @ S
//   attn  = mask_{i>=j}( (q e^{gc_i}) . k_j * e^{-gc_j} )
//   out   = q' @ S + attn @ v_new
//   S     = e^{g_tail} S + (k e^{g_tail-gc})^T @ v_new
// block 256 (8 waves)
// ---------------------------------------------------------------------------
__global__ void core_kernel(const _Float16* __restrict__ qh,
                            const _Float16* __restrict__ kh,
                            const _Float16* __restrict__ wbuf,
                            const _Float16* __restrict__ ubuf,
                            const float* __restrict__ g,
                            float* __restrict__ obuf) {
  __shared__ float    sS[DKc * VBc];    // state slice, f32
  __shared__ _Float16 sP[BTc * DKc];    // -w chunk, then q' chunk
  __shared__ _Float16 sK[BTc * DKc];    // k chunk
  __shared__ _Float16 sattn[BTc * BTc];
  __shared__ _Float16 snew[BTc * VBc];
  __shared__ float    sgl[BTc], sgc[BTc];
  __shared__ float    segc[BTc], sinv[BTc], sekg[BTc];

  const int bh = blockIdx.x >> 2, vb0 = (blockIdx.x & 3) * VBc;
  const int tid = threadIdx.x, lane = tid & 31, wid = tid >> 5;
  const int m = lane & 15, hh = lane >> 4;

  for (int idx = tid; idx < DKc * VBc; idx += 256) sS[idx] = 0.0f;
  __syncthreads();

  for (int nc = 0; nc < NTc; ++nc) {
    const int t0 = nc * BTc;
    const size_t base = ((size_t)bh * Tc + t0) * DKc;

    // Prefetch next chunk's tiles into cache while this chunk computes
    // (global_prefetch_b8; one 128B cacheline per thread, 128 lines/buffer).
    if (nc + 1 < NTc && tid < 128) {
      const size_t bn = base + (size_t)BTc * DKc + (size_t)tid * 64;
      __builtin_prefetch(wbuf + bn, 0, 1);
      __builtin_prefetch(ubuf + bn, 0, 1);
      __builtin_prefetch(qh + bn, 0, 1);
      __builtin_prefetch(kh + bn, 0, 1);
    }

    if (tid < BTc) sgl[tid] = g[(size_t)bh * Tc + t0 + tid];
    __syncthreads();
    if (tid < BTc) {
      float s = 0.f;
      for (int j = 0; j <= tid; ++j) s += sgl[j];
      sgc[tid] = s;
    }
    __syncthreads();
    const float gtail = sgc[BTc - 1];
    if (tid < BTc) {
      const float s = sgc[tid];
      segc[tid] = __expf(s);
      sinv[tid] = __expf(-s);
      sekg[tid] = __expf(gtail - s);
    }

    // stage -w
    for (int idx = tid; idx < BTc * DKc; idx += 256)
      sP[idx] = (_Float16)(-(float)wbuf[base + idx]);
    __syncthreads();

    // v_new = u + (-w) @ S   (8 tiles; 1 per wave)
    {
      const int i0 = (wid >> 1) * 16, n0 = (wid & 1) * 16;
      v8f c;
#pragma unroll
      for (int r = 0; r < 8; ++r)
        c[r] = (float)ubuf[((size_t)bh * Tc + t0 + i0 + r + 8 * hh) * DVc +
                           vb0 + n0 + m];
#pragma unroll
      for (int ks = 0; ks < 4; ++ks) {
        v16h a = load_a_rm(sP, DKc, i0, ks * 32, m, hh);
        v16h b = load_b_km_f32(sS, VBc, ks * 32, n0, m, hh);
        c = wmma16(a, b, c);
      }
#pragma unroll
      for (int r = 0; r < 8; ++r)
        snew[(i0 + r + 8 * hh) * VBc + n0 + m] = (_Float16)c[r];
    }
    __syncthreads();

    // stage q' = q*exp(gc) (into sP) and raw k (into sK)
    for (int idx = tid; idx < BTc * DKc; idx += 256) {
      const int tl = idx >> 7;
      sP[idx] = (_Float16)((float)qh[base + idx] * segc[tl]);
      sK[idx] = kh[base + idx];
    }
    __syncthreads();

    // attn (16 tiles; 2 per wave): (q' @ k^T) * e^{-gc_j}, mask i>=j
    for (int rep = 0; rep < 2; ++rep) {
      const int tile = wid + 8 * rep;
      const int i0 = (tile >> 2) * 16, j0 = (tile & 3) * 16;
      v8f c = {};
#pragma unroll
      for (int ks = 0; ks < 4; ++ks) {
        v16h a = load_a_rm(sP, DKc, i0, ks * 32, m, hh);
        v16h b = load_b_nm(sK, DKc, j0, ks * 32, m, hh);
        c = wmma16(a, b, c);
      }
#pragma unroll
      for (int r = 0; r < 8; ++r) {
        const int i = i0 + r + 8 * hh, j = j0 + m;
        const float v = c[r] * sinv[j];
        sattn[i * BTc + j] = (_Float16)((i >= j) ? v : 0.0f);
      }
    }
    __syncthreads();

    // out = q' @ S + attn @ v_new (8 tiles; 1 per wave)
    {
      const int i0 = (wid >> 1) * 16, n0 = (wid & 1) * 16;
      v8f c = {};
#pragma unroll
      for (int ks = 0; ks < 4; ++ks) {
        v16h a = load_a_rm(sP, DKc, i0, ks * 32, m, hh);
        v16h b = load_b_km_f32(sS, VBc, ks * 32, n0, m, hh);
        c = wmma16(a, b, c);
      }
#pragma unroll
      for (int ks = 0; ks < 2; ++ks) {
        v16h a = load_a_rm(sattn, BTc, i0, ks * 32, m, hh);
        v16h b = load_b_km(snew, VBc, ks * 32, n0, m, hh);
        c = wmma16(a, b, c);
      }
#pragma unroll
      for (int r = 0; r < 8; ++r)
        obuf[((size_t)bh * Tc + t0 + i0 + r + 8 * hh) * DVc + vb0 + n0 + m] =
            c[r];
    }
    __syncthreads();

    // S = e^{gtail} S + kg^T @ v_new (16 tiles; 2 per wave)
    const float eg = __expf(gtail);
    for (int rep = 0; rep < 2; ++rep) {
      const int tile = wid + 8 * rep;
      const int m0 = (tile >> 1) * 16, n0 = (tile & 1) * 16;
      v8f c;
#pragma unroll
      for (int r = 0; r < 8; ++r)
        c[r] = sS[(m0 + r + 8 * hh) * VBc + n0 + m] * eg;
#pragma unroll
      for (int ks = 0; ks < 2; ++ks) {
        const int k0 = ks * 32;
        v16h a;  // A[M=dk][K=j] = k[j][dk] * exp(gtail - gc[j])
#pragma unroll
        for (int e = 0; e < 16; ++e) {
          const int j = k0 + kmap(e, hh);
          a[e] = (_Float16)((float)sK[j * DKc + m0 + m] * sekg[j]);
        }
        v16h b = load_b_km(snew, VBc, k0, n0, m, hh);
        c = wmma16(a, b, c);
      }
#pragma unroll
      for (int r = 0; r < 8; ++r)
        sS[(m0 + r + 8 * hh) * VBc + n0 + m] = c[r];
    }
    __syncthreads();
  }
}

// ---------------------------------------------------------------------------
// Kernel 4: gated RMSNorm over DV + output layout (B,T,H*DV), f32 out.
// grid = B*T*H, block 128
// ---------------------------------------------------------------------------
__global__ void norm_kernel(const float* __restrict__ obuf,
                            const float* __restrict__ gate,
                            const float* __restrict__ nw,
                            float* __restrict__ out) {
  __shared__ float red[DVc];
  const int h = blockIdx.x % Hc;
  const int bt = blockIdx.x / Hc;
  const int b = bt / Tc, t = bt % Tc;
  const int d = threadIdx.x;
  const float x = obuf[((size_t)(b * Hc + h) * Tc + t) * DVc + d];
  red[d] = x * x;
  __syncthreads();
  for (int s = DVc / 2; s > 0; s >>= 1) {
    if (d < s) red[d] += red[d + s];
    __syncthreads();
  }
  const float inv = rsqrtf(red[0] / (float)DVc + EPSc);
  const size_t gidx = ((size_t)b * Tc + t) * (Hc * DVc) + h * DVc + d;
  const float gt = gate[gidx];
  out[gidx] = x * inv * nw[d] * (gt * sigmoidf_(gt));
}

// ---------------------------------------------------------------------------
extern "C" void kernel_launch(void* const* d_in, const int* in_sizes, int n_in,
                              void* d_out, int out_size, void* d_ws,
                              size_t ws_size, hipStream_t stream) {
  (void)in_sizes; (void)n_in; (void)out_size; (void)ws_size;
  const float* q_pre = (const float*)d_in[0];
  const float* k_pre = (const float*)d_in[1];
  const float* v_pre = (const float*)d_in[2];
  const float* cwq   = (const float*)d_in[3];
  const float* cbq   = (const float*)d_in[4];
  const float* cwk   = (const float*)d_in[5];
  const float* cbk   = (const float*)d_in[6];
  const float* cwv   = (const float*)d_in[7];
  const float* cbv   = (const float*)d_in[8];
  const float* g     = (const float*)d_in[9];
  const float* beta  = (const float*)d_in[10];
  const float* gate  = (const float*)d_in[11];
  const float* nw    = (const float*)d_in[12];
  float* out = (float*)d_out;

  const size_t NE = (size_t)Bc * Hc * Tc * DKc;  // 16.7M elements
  char* ws = (char*)d_ws;
  _Float16* qh   = (_Float16*)ws;
  _Float16* kh   = qh + NE;
  _Float16* vh   = kh + NE;
  _Float16* wbuf = vh + NE;
  _Float16* ubuf = wbuf + NE;
  float*    obuf = (float*)(ubuf + NE);

  dim3 gconv(Bc * Hc, NTc);
  conv_silu_kernel<<<gconv, 256, 0, stream>>>(q_pre, cwq, cbq, qh, 0);
  conv_silu_kernel<<<gconv, 256, 0, stream>>>(k_pre, cwk, cbk, kh, 1);
  conv_silu_kernel<<<gconv, 256, 0, stream>>>(v_pre, cwv, cbv, vh, 2);

  prep_kernel<<<Bc * Hc * NTc, 256, 0, stream>>>(kh, vh, g, beta, wbuf, ubuf);

  core_kernel<<<Bc * Hc * (DVc / VBc), 256, 0, stream>>>(qh, kh, wbuf, ubuf, g,
                                                         obuf);

  norm_kernel<<<Bc * Tc * Hc, DVc, 0, stream>>>(obuf, gate, nw, out);
}
